// AttentionAggregator_82102594830566
// MI455X (gfx1250) — matrix operand order repaired and checked by reading, not compiled, not run on hardware
//
#include <hip/hip_runtime.h>
#include <hip/hip_bf16.h>

// ---------------------------------------------------------------------------
// AttentionAggregator for MI455X (gfx1250, wave32, WMMA + double-buffered
// async global->LDS DMA).  B=4, S=2048, H=1024, heads=16, dk=64.
// ---------------------------------------------------------------------------

typedef __bf16 bf16_t;
typedef __attribute__((ext_vector_type(16))) __bf16 v16bf;
typedef __attribute__((ext_vector_type(8)))  __bf16 v8bf;
typedef __attribute__((ext_vector_type(8)))  float  v8f;

#define WMMA_BF16(A, B, C) \
  __builtin_amdgcn_wmma_f32_16x16x32_bf16(false, (A), false, (B), (short)0, (C), false, false)

static constexpr int Bq   = 4;
static constexpr int S    = 2048;
static constexpr int H    = 1024;
static constexpr int NH   = 16;
static constexpr int DK   = 64;
static constexpr int LDSP = 40;           // padded row stride (bf16) for 32-wide panels
static constexpr int LDKP = 72;           // padded row stride (bf16) for 64-wide K panel
static constexpr float NEG_FILL = -10000.0f;

__device__ __forceinline__ v16bf cat8(v8bf lo, v8bf hi) {
  return __builtin_shufflevector(lo, hi, 0,1,2,3,4,5,6,7,8,9,10,11,12,13,14,15);
}
// A-operand fragment (16x32 bf16): lane half h: elems = K[8h..8h+7] ++ K[8h+16..8h+23]
__device__ __forceinline__ v16bf ldA(const bf16_t* p, int kb8) {
  v8bf lo = *(const v8bf*)(p + kb8);
  v8bf hi = *(const v8bf*)(p + kb8 + 16);
  return cat8(lo, hi);
}
// B-operand fragment (32x16 bf16): lane half h: elems = K[8h..8h+7] ++ K[8h+16..8h+23]
__device__ __forceinline__ v16bf ldB(const bf16_t* p, int kb16) {
  v8bf lo = *(const v8bf*)(p + kb16 / 2);
  v8bf hi = *(const v8bf*)(p + kb16 / 2 + 16);
  return cat8(lo, hi);
}

// gfx1250 async DMA: global -> LDS, 16B per lane, tracked by ASYNCcnt.
__device__ __forceinline__ void async_load_b128(unsigned lds_byte_off, const void* gaddr) {
  asm volatile("global_load_async_to_lds_b128 %0, %1, off"
               :: "v"(lds_byte_off), "v"((unsigned long long)(uintptr_t)gaddr)
               : "memory");
}
__device__ __forceinline__ void wait_async0() {
  asm volatile("s_wait_asynccnt 0x0" ::: "memory");
}
__device__ __forceinline__ void wait_async2() {   // keep 2 newest in flight
  asm volatile("s_wait_asynccnt 0x2" ::: "memory");
}
__device__ __forceinline__ void wait_async4() {   // keep 4 newest in flight
  asm volatile("s_wait_asynccnt 0x4" ::: "memory");
}
__device__ __forceinline__ unsigned lds_off(const void* p) {
  return (unsigned)(uintptr_t)p;          // LDS aperture: addr[31:0] == LDS byte offset
}

// ---------------------------------------------------------------------------
// Kernel 0: elementwise f32 -> bf16 conversion (x, Wq, Wk, Wv pre-pass)
// ---------------------------------------------------------------------------
__global__ __launch_bounds__(256)
void cvt_bf16_kernel(const float* __restrict__ src, bf16_t* __restrict__ dst, int n8)
{
  const int idx = blockIdx.x * 256 + threadIdx.x;
  if (idx >= n8) return;
  const float4* s = (const float4*)(src + (size_t)idx * 8);
  const float4 a = s[0], b = s[1];
  v8bf o = {(bf16_t)a.x,(bf16_t)a.y,(bf16_t)a.z,(bf16_t)a.w,
            (bf16_t)b.x,(bf16_t)b.y,(bf16_t)b.z,(bf16_t)b.w};
  *(v8bf*)(dst + (size_t)idx * 8) = o;
}

// ---------------------------------------------------------------------------
// Kernel 1: fused QKV projection.  y = xb @ Wb^T (+bias for V), bf16 -> bf16.
//   Q,K stored (b,h,S,dk) row-major; V stored transposed (b,h,dk,S).
//   K-panels (128x32 each for A and B) streamed by double-buffered async DMA.
// ---------------------------------------------------------------------------
__global__ __launch_bounds__(256)
void qkv_kernel(const bf16_t* __restrict__ xb,  const bf16_t* __restrict__ Wqb,
                const bf16_t* __restrict__ Wkb, const bf16_t* __restrict__ Wvb,
                const float* __restrict__ bv,
                bf16_t* __restrict__ Qb, bf16_t* __restrict__ Kb, bf16_t* __restrict__ Vt)
{
  __shared__ bf16_t lA[2][128 * LDSP];
  __shared__ bf16_t lB[2][128 * LDSP];

  const int proj = blockIdx.z;                       // 0=Q 1=K 2=V
  const bf16_t* __restrict__ W = (proj == 0) ? Wqb : ((proj == 1) ? Wkb : Wvb);

  const int t    = threadIdx.x;
  const int lane = t & 31;
  const int wave = t >> 5;
  const int half = lane >> 4;
  const int l15  = lane & 15;
  const int kb8  = half * 8;
  const int kb16 = half * 16;
  const int mbase = (wave >> 2) * 64;
  const int nbase = (wave & 3) * 32;
  const int m0 = blockIdx.x * 128;                   // global flattened (b,s) row
  const int n0 = blockIdx.y * 128;                   // global output feature

  // async staging: each thread DMAs 2 rows per matrix (16B each)
  const int r0 = t >> 2;                             // 0..63
  const int c0 = (t & 3) * 8;                        // 0,8,16,24 (bf16 elems)

  auto issueAB = [&](int buf, int kc) {
    async_load_b128(lds_off(&lA[buf][ r0       * LDSP + c0]), xb + (size_t)(m0 + r0     ) * H + kc + c0);
    async_load_b128(lds_off(&lA[buf][(r0 + 64) * LDSP + c0]), xb + (size_t)(m0 + r0 + 64) * H + kc + c0);
    async_load_b128(lds_off(&lB[buf][ r0       * LDSP + c0]), W  + (size_t)(n0 + r0     ) * H + kc + c0);
    async_load_b128(lds_off(&lB[buf][(r0 + 64) * LDSP + c0]), W  + (size_t)(n0 + r0 + 64) * H + kc + c0);
  };

  v8f acc[4][2];
#pragma unroll
  for (int i = 0; i < 4; ++i)
#pragma unroll
    for (int j = 0; j < 2; ++j) acc[i][j] = (v8f)0.0f;

  issueAB(0, 0);
  for (int kc = 0, it = 0; kc < H; kc += 32, ++it) {
    const int cur = it & 1;
    __syncthreads();                                 // prev compute done everywhere
    const bool more = (kc + 32) < H;
    if (more) { issueAB(cur ^ 1, kc + 32); wait_async4(); }
    else      { wait_async0(); }
    __syncthreads();                                 // DMA of buf[cur] visible

    v16bf aF[4], bF[2];
#pragma unroll
    for (int i = 0; i < 4; ++i)
      aF[i] = ldA(&lA[cur][(mbase + i * 16 + l15) * LDSP], kb8);
#pragma unroll
    for (int j = 0; j < 2; ++j)
      bF[j] = ldB(&lB[cur][(nbase + j * 16 + l15) * LDSP], kb16);
#pragma unroll
    for (int i = 0; i < 4; ++i)
#pragma unroll
      for (int j = 0; j < 2; ++j)
        acc[i][j] = WMMA_BF16(aF[i], bF[j], acc[i][j]);
  }

  // store: C layout lane holds N = l15, M = r + 8*half per v8f slot r
#pragma unroll
  for (int j = 0; j < 2; ++j) {
    const int col  = n0 + nbase + j * 16 + l15;      // output feature 0..1023
    const int head = col >> 6, d = col & 63;
    const float bvv = (proj == 2) ? bv[col] : 0.0f;
#pragma unroll
    for (int i = 0; i < 4; ++i) {
      if (proj == 2) {
        // transposed V: lane's 8 rows are 8 consecutive s -> one 16B store
        const int sbase = m0 + mbase + i * 16 + half * 8;
        const int b = sbase >> 11, s = sbase & (S - 1);
        const size_t bh = (size_t)(b * NH + head);
        v8f a = acc[i][j];
        v8bf pv = {(bf16_t)(a[0]+bvv),(bf16_t)(a[1]+bvv),(bf16_t)(a[2]+bvv),(bf16_t)(a[3]+bvv),
                   (bf16_t)(a[4]+bvv),(bf16_t)(a[5]+bvv),(bf16_t)(a[6]+bvv),(bf16_t)(a[7]+bvv)};
        *(v8bf*)(Vt + (bh * DK + d) * S + s) = pv;
      } else {
#pragma unroll
        for (int r = 0; r < 8; ++r) {
          const int grow = m0 + mbase + i * 16 + r + half * 8;
          const int b = grow >> 11, s = grow & (S - 1);
          const size_t bh = (size_t)(b * NH + head);
          const float v = acc[i][j][r];
          if (proj == 0) Qb[(bh * S + s) * DK + d] = (bf16_t)v;
          else           Kb[(bh * S + s) * DK + d] = (bf16_t)v;
        }
      }
    }
  }
}

// ---------------------------------------------------------------------------
// Kernel 2: flash attention + masked row-sum of ctx.
//   Block = 8 waves = 8 query tiles of one (b,head).  K (32x64) and V (64x32)
//   chunks streamed into double-buffered LDS via async DMA; all waves share.
// ---------------------------------------------------------------------------
__global__ __launch_bounds__(256)
void attn_kernel(const bf16_t* __restrict__ Qb, const bf16_t* __restrict__ Kb,
                 const bf16_t* __restrict__ Vt, const int* __restrict__ L,
                 float* __restrict__ partials)
{
  __shared__ bf16_t lK[2][32 * LDKP];                // 2 x 4.5 KB
  __shared__ bf16_t lV[2][64 * LDSP];                // 2 x 5 KB
  __shared__ bf16_t lP[8][16 * LDSP];                // per-wave P staging

  const int t    = threadIdx.x;
  const int lane = t & 31;
  const int wave = t >> 5;
  const int half = lane >> 4;
  const int l15  = lane & 15;
  const int kb8  = half * 8;
  const int kb16 = half * 16;

  const int b = blockIdx.z, h = blockIdx.y;
  const int qtile = blockIdx.x * 8 + wave;
  const int qbase = qtile * 16;
  const size_t bh = (size_t)(b * NH + h);
  const int Lb = L[b];

  const bf16_t* __restrict__ Qp = Qb + bh * S * DK;
  const bf16_t* __restrict__ Kp = Kb + bh * S * DK;
  const bf16_t* __restrict__ Vp = Vt + bh * DK * S;
  bf16_t* myP = &lP[wave][0];

  // staging assignment: one async b128 per thread per tile
  const int krow = t >> 3, kcol = (t & 7) * 8;       // K: 32 rows x 64
  const int vrow = t >> 2, vcol = (t & 3) * 8;       // V: 64 rows x 32

  auto issueKV = [&](int buf, int key0) {
    async_load_b128(lds_off(&lK[buf][krow * LDKP + kcol]), Kp + (key0 + krow) * DK + kcol);
    async_load_b128(lds_off(&lV[buf][vrow * LDSP + vcol]), Vp + (size_t)vrow * S + key0 + vcol);
  };

  // Q fragments (16 rows x 64, two 32-wide K steps), loaded once
  v16bf qF[2];
#pragma unroll
  for (int ks = 0; ks < 2; ++ks)
    qF[ks] = ldA(Qp + (qbase + l15) * DK + ks * 32, kb8);

  float mrun[8], lrun[8];
  v8f ctx[4];
#pragma unroll
  for (int r = 0; r < 8; ++r) { mrun[r] = -3.0e38f; lrun[r] = 0.0f; }
#pragma unroll
  for (int dt = 0; dt < 4; ++dt) ctx[dt] = (v8f)0.0f;

  const int nch = (Lb + 31) >> 5;                    // chunks of 32 keys, uniform
  issueKV(0, 0);
  for (int c = 0; c < nch; ++c) {
    const int cur = c & 1;
    const int key0 = c * 32;
    __syncthreads();                                 // prev compute done everywhere
    if (c + 1 < nch) { issueKV(cur ^ 1, key0 + 32); wait_async2(); }
    else             { wait_async0(); }
    __syncthreads();                                 // DMA of buf[cur] visible

    // ---- scores: S = Q K^T / 8, two 16-key column tiles ----
    v8f sc[2];
    sc[0] = (v8f)0.0f; sc[1] = (v8f)0.0f;
#pragma unroll
    for (int kt = 0; kt < 2; ++kt) {
      const bf16_t* kr = &lK[cur][(kt * 16 + l15) * LDKP];
      v16bf kF0 = ldB(kr, kb16);
      v16bf kF1 = ldB(kr + 32, kb16);
      sc[kt] = WMMA_BF16(qF[0], kF0, sc[kt]);
      sc[kt] = WMMA_BF16(qF[1], kF1, sc[kt]);
    }
    // scale + key mask (per-lane column)
#pragma unroll
    for (int kt = 0; kt < 2; ++kt) {
      const bool valid = (key0 + kt * 16 + l15) < Lb;
#pragma unroll
      for (int r = 0; r < 8; ++r) {
        const float v = sc[kt][r] * 0.125f;
        sc[kt][r] = valid ? v : NEG_FILL;
      }
    }
    // ---- online softmax ----
    float rv[8];
#pragma unroll
    for (int r = 0; r < 8; ++r) rv[r] = fmaxf(sc[0][r], sc[1][r]);
#pragma unroll
    for (int off = 1; off < 16; off <<= 1)
#pragma unroll
      for (int r = 0; r < 8; ++r) rv[r] = fmaxf(rv[r], __shfl_xor(rv[r], off, 32));

    float alpha[8], rs[8];
#pragma unroll
    for (int r = 0; r < 8; ++r) {
      const float mnew = fmaxf(mrun[r], rv[r]);
      alpha[r] = __expf(mrun[r] - mnew);
      mrun[r] = mnew;
      const float p0 = __expf(sc[0][r] - mnew);
      const float p1 = __expf(sc[1][r] - mnew);
      sc[0][r] = p0; sc[1][r] = p1;
      rs[r] = p0 + p1;
    }
#pragma unroll
    for (int off = 1; off < 16; off <<= 1)
#pragma unroll
      for (int r = 0; r < 8; ++r) rs[r] += __shfl_xor(rs[r], off, 32);
#pragma unroll
    for (int r = 0; r < 8; ++r) lrun[r] = lrun[r] * alpha[r] + rs[r];
#pragma unroll
    for (int dt = 0; dt < 4; ++dt)
#pragma unroll
      for (int r = 0; r < 8; ++r) ctx[dt][r] *= alpha[r];

    // ---- C-layout P -> per-wave LDS -> A-layout fragment ----
#pragma unroll
    for (int kt = 0; kt < 2; ++kt)
#pragma unroll
      for (int r = 0; r < 8; ++r)
        myP[(r + half * 8) * LDSP + kt * 16 + l15] = (bf16_t)sc[kt][r];
    const v16bf pF = ldA(myP + l15 * LDSP, kb8);      // same-wave DS RAW (dscnt)

    // ---- ctx += P @ V, four 16-dim tiles from shared V ----
#pragma unroll
    for (int dt = 0; dt < 4; ++dt) {
      v16bf vF = ldB(&lV[cur][(dt * 16 + l15) * LDSP], kb16);
      ctx[dt] = WMMA_BF16(pF, vF, ctx[dt]);
    }
  }

  // epilogue: ctx /= l, mask invalid query rows, reduce 16 rows -> partials
  float w[8];
#pragma unroll
  for (int r = 0; r < 8; ++r) {
    const int s = qbase + r + half * 8;
    const float inv = (lrun[r] > 0.0f) ? (1.0f / lrun[r]) : 0.0f;
    w[r] = (s < Lb) ? inv : 0.0f;
  }
#pragma unroll
  for (int dt = 0; dt < 4; ++dt) {
    float a = 0.0f;
#pragma unroll
    for (int r = 0; r < 8; ++r) a += ctx[dt][r] * w[r];
    a += __shfl_xor(a, 16, 32);
    if (half == 0)
      partials[(bh * (S / 16) + qtile) * DK + dt * 16 + l15] = a;
  }
}

// ---------------------------------------------------------------------------
// Kernel 3: pooled = (sum_tiles partials) @ Wo^T / L + bo   (4 x 1024 GEMV)
// ---------------------------------------------------------------------------
__global__ __launch_bounds__(256)
void pool_proj_kernel(const float* __restrict__ partials, const float* __restrict__ Wo,
                      const float* __restrict__ bo, const int* __restrict__ L,
                      float* __restrict__ out)
{
  __shared__ float vec[H];
  const int t = threadIdx.x;
  const int b = blockIdx.y;
#pragma unroll
  for (int i = 0; i < 4; ++i) {
    const int hf = i * 256 + t;
    const int head = hf >> 6, d = hf & 63;
    const float* p = partials + ((size_t)(b * NH + head) * (S / 16)) * DK + d;
    float s = 0.0f;
    for (int tile = 0; tile < S / 16; ++tile) s += p[tile * DK];
    vec[hf] = s;
  }
  __syncthreads();
  const int o = blockIdx.x * 256 + t;
  const float4* wrow = (const float4*)(Wo + (size_t)o * H);
  float acc = 0.0f;
  for (int hq = 0; hq < H / 4; ++hq) {
    const float4 wv = wrow[hq];
    const float4 vv = *(const float4*)&vec[hq * 4];
    acc += wv.x * vv.x + wv.y * vv.y + wv.z * vv.z + wv.w * vv.w;
  }
  out[(size_t)b * H + o] = acc / (float)L[b] + bo[o];
}

// ---------------------------------------------------------------------------
extern "C" void kernel_launch(void* const* d_in, const int* in_sizes, int n_in,
                              void* d_out, int out_size, void* d_ws, size_t ws_size,
                              hipStream_t stream)
{
  const float* x  = (const float*)d_in[0];
  const int*   L  = (const int*)  d_in[1];
  const float* Wq = (const float*)d_in[2];
  const float* Wk = (const float*)d_in[3];
  const float* Wv = (const float*)d_in[4];
  const float* bv = (const float*)d_in[5];
  const float* Wo = (const float*)d_in[6];
  const float* bo = (const float*)d_in[7];
  float* out = (float*)d_out;

  char* ws = (char*)d_ws;
  const size_t qkv_bytes = (size_t)Bq * NH * S * DK * sizeof(bf16_t);   // 16 MB each
  const size_t w_bytes   = (size_t)H * H * sizeof(bf16_t);              // 2 MB each
  bf16_t* Qb = (bf16_t*)(ws + 0 * qkv_bytes);
  bf16_t* Kb = (bf16_t*)(ws + 1 * qkv_bytes);
  bf16_t* Vt = (bf16_t*)(ws + 2 * qkv_bytes);
  float* partials = (float*)(ws + 3 * qkv_bytes);                       // 2 MB
  char* p = ws + 3 * qkv_bytes + (size_t)Bq * NH * (S / 16) * DK * sizeof(float);
  bf16_t* xb  = (bf16_t*)(p);                                           // 16 MB
  bf16_t* Wqb = (bf16_t*)(p + qkv_bytes);
  bf16_t* Wkb = (bf16_t*)(p + qkv_bytes + 1 * w_bytes);
  bf16_t* Wvb = (bf16_t*)(p + qkv_bytes + 2 * w_bytes);

  // pre-pass: f32 -> bf16 for activations and QKV weights
  const int nx8 = (Bq * S * H) / 8, nw8 = (H * H) / 8;
  cvt_bf16_kernel<<<dim3((nx8 + 255) / 256), 256, 0, stream>>>(x,  xb,  nx8);
  cvt_bf16_kernel<<<dim3((nw8 + 255) / 256), 256, 0, stream>>>(Wq, Wqb, nw8);
  cvt_bf16_kernel<<<dim3((nw8 + 255) / 256), 256, 0, stream>>>(Wk, Wkb, nw8);
  cvt_bf16_kernel<<<dim3((nw8 + 255) / 256), 256, 0, stream>>>(Wv, Wvb, nw8);

  dim3 g1((Bq * S) / 128, H / 128, 3);
  qkv_kernel<<<g1, 256, 0, stream>>>(xb, Wqb, Wkb, Wvb, bv, Qb, Kb, Vt);

  dim3 g2(S / (16 * 8), NH, Bq);
  attn_kernel<<<g2, 256, 0, stream>>>(Qb, Kb, Vt, L, partials);

  dim3 g3(H / 256, Bq);
  pool_proj_kernel<<<g3, 256, 0, stream>>>(partials, Wo, bo, L, out);
}
